// LinearAttention_51771535786580
// MI455X (gfx1250) — matrix operand order repaired and checked by reading, not compiled, not run on hardware
//
#include <hip/hip_runtime.h>

typedef __bf16 bf16_t;
typedef __attribute__((ext_vector_type(16))) __bf16 v16bf;
typedef __attribute__((ext_vector_type(8)))  __bf16 bf16x8;
typedef __attribute__((ext_vector_type(4)))  __bf16 bf16x4;
typedef __attribute__((ext_vector_type(8)))  float  v8f;

#define DMODEL 1024
#define NHEADS 16
#define DHEAD  64
#define BATCH  2
#define SEQ    2048
#define MTOT   (BATCH * SEQ)   // 4096 tokens

// ---------------------------------------------------------------- helpers
__device__ __forceinline__ v8f zero8() {
  v8f z;
#pragma unroll
  for (int i = 0; i < 8; ++i) z[i] = 0.0f;
  return z;
}

// B-operand fragment: 16 contiguous K values per lane (two 16B loads).
__device__ __forceinline__ v16bf load_contig16(const bf16_t* p) {
  bf16x8 lo = *(const bf16x8*)p;
  bf16x8 hi = *(const bf16x8*)(p + 8);
  v16bf r;
#pragma unroll
  for (int i = 0; i < 8; ++i) { r[i] = lo[i]; r[i + 8] = hi[i]; }
  return r;
}

// A-operand fragment: 16 contiguous K values relative to p (two 16B loads).
__device__ __forceinline__ v16bf load_afrag(const bf16_t* p) {
  bf16x8 lo = *(const bf16x8*)p;
  bf16x8 hi = *(const bf16x8*)(p + 8);
  v16bf r;
#pragma unroll
  for (int i = 0; i < 8; ++i) { r[i] = lo[i]; r[i + 8] = hi[i]; }
  return r;
}

// Async global -> LDS copy of 16 bytes (CDNA5 GLOBAL_LOAD_ASYNC_TO_LDS_B128,
// tracked by ASYNCcnt). lds_off = byte offset in LDS (low 32 bits of the
// generic shared-space address), g = per-lane global address.
__device__ __forceinline__ void async_cp16(unsigned lds_off, const bf16_t* g) {
  asm volatile("global_load_async_to_lds_b128 %0, %1, off"
               :: "v"(lds_off), "v"(g)
               : "memory");
}

// ---------------------------------------------------------------- fp32 -> bf16
__global__ __launch_bounds__(256)
void cvt_f32_bf16(const float4* __restrict__ s, bf16x4* __restrict__ d, int n4) {
  int i = blockIdx.x * blockDim.x + threadIdx.x;
  if (i < n4) {
    float4 v = s[i];
    bf16x4 o;
    o[0] = (bf16_t)v.x; o[1] = (bf16_t)v.y; o[2] = (bf16_t)v.z; o[3] = (bf16_t)v.w;
    d[i] = o;
  }
}

// ---------------------------------------------------------------- GEMM
// C[M=4096,N=1024] = A[M,K=1024] * W[N,K]^T + bias[N]
// Double-buffered async global->LDS staging overlapped with WMMA.
// MODE 0: write bf16 per-head Q/K layout   [b][h][l][64]
// MODE 1: write bf16 per-head V transposed [b][h][64][l]
// MODE 2: write fp32 row-major [M][N] (final output)
template <int MODE>
__global__ __launch_bounds__(256)
void gemm_bf16(const bf16_t* __restrict__ A, const bf16_t* __restrict__ W,
               const float* __restrict__ bias, void* __restrict__ outp) {
  __shared__ bf16_t As[2][128 * 40];   // 128x32 tile, pitch 40 (pad), 2 buffers
  __shared__ bf16_t Bs[2][128 * 40];

  const int t    = threadIdx.x;
  const int lane = t & 31;
  const int wave = t >> 5;          // 8 waves
  const int wm   = wave >> 1;       // 0..3 -> 32-row strip
  const int wn   = wave & 1;        // 0..1 -> 64-col strip
  const int l15  = lane & 15;
  const int hi   = (lane >= 16);
  const int aoff = hi ? 16 : 0;
  const int boff = hi ? 16 : 0;
  const int rb   = hi ? 8 : 0;

  const int mBase = blockIdx.y * 128;
  const int nBase = blockIdx.x * 128;

  const int sr = t >> 1;            // staging row 0..127
  const int sh = (t & 1) * 16;      // staging half (elements)

  const bf16_t* gA = A + (size_t)(mBase + sr) * DMODEL + sh;
  const bf16_t* gB = W + (size_t)(nBase + sr) * DMODEL + sh;
  const unsigned ldsA0 = (unsigned)(uintptr_t)(&As[0][sr * 40 + sh]);
  const unsigned ldsB0 = (unsigned)(uintptr_t)(&Bs[0][sr * 40 + sh]);
  const unsigned bufStride = (unsigned)(128 * 40 * sizeof(bf16_t));

  v8f acc[2][4];
#pragma unroll
  for (int a = 0; a < 2; ++a)
#pragma unroll
    for (int b = 0; b < 4; ++b) acc[a][b] = zero8();

  // prologue: stage tile 0 into buffer 0 (4 async ops per wave)
  async_cp16(ldsA0,      gA);
  async_cp16(ldsA0 + 16, gA + 8);
  async_cp16(ldsB0,      gB);
  async_cp16(ldsB0 + 16, gB + 8);

  const int NK = DMODEL / 32;
  for (int kt = 0; kt < NK; ++kt) {
    const int cur = kt & 1;
    if (kt + 1 < NK) {
      // stage next tile into the other buffer while we compute this one
      const unsigned nb = (unsigned)(cur ^ 1) * bufStride;
      const bf16_t* na = gA + (kt + 1) * 32;
      const bf16_t* nw = gB + (kt + 1) * 32;
      async_cp16(ldsA0 + nb,      na);
      async_cp16(ldsA0 + nb + 16, na + 8);
      async_cp16(ldsB0 + nb,      nw);
      async_cp16(ldsB0 + nb + 16, nw + 8);
      // first 4 outstanding async ops (current tile) must have landed
      asm volatile("s_wait_asynccnt 0x4" ::: "memory");
    } else {
      asm volatile("s_wait_asynccnt 0x0" ::: "memory");
    }
    __syncthreads();

    v16bf af[2];
#pragma unroll
    for (int m2 = 0; m2 < 2; ++m2)
      af[m2] = load_afrag(&As[cur][(wm * 32 + m2 * 16 + l15) * 40 + aoff]);

#pragma unroll
    for (int i = 0; i < 4; ++i) {
      v16bf bfv = load_contig16(&Bs[cur][(wn * 64 + i * 16 + l15) * 40 + boff]);
#pragma unroll
      for (int m2 = 0; m2 < 2; ++m2)
        acc[m2][i] = __builtin_amdgcn_wmma_f32_16x16x32_bf16(
            false, af[m2], false, bfv, (short)0, acc[m2][i], false, false);
    }
    __syncthreads();
  }

  // epilogue: bias + layout-specific store
#pragma unroll
  for (int i = 0; i < 4; ++i) {
    const int n = nBase + wn * 64 + i * 16 + l15;
    const float bv = bias[n];
#pragma unroll
    for (int m2 = 0; m2 < 2; ++m2) {
#pragma unroll
      for (int r = 0; r < 8; ++r) {
        const int m = mBase + wm * 32 + m2 * 16 + rb + r;
        const float v = acc[m2][i][r] + bv;
        if (MODE == 2) {
          ((float*)outp)[(size_t)m * DMODEL + n] = v;
        } else {
          const bf16_t bo = (bf16_t)v;
          const int bb = m >> 11, l = m & 2047, h = n >> 6, d = n & 63;
          if (MODE == 0)
            ((bf16_t*)outp)[((size_t)((bb * NHEADS + h) * SEQ + l)) * DHEAD + d] = bo;
          else
            ((bf16_t*)outp)[((size_t)((bb * NHEADS + h) * DHEAD + d)) * SEQ + l] = bo;
        }
      }
    }
  }
}

// ---------------------------------------------------------------- causal linear attention
// Per block: one (b,h), one 128-row band of l. out = tril(Q K^T) V.
__global__ __launch_bounds__(256)
void linattn(const bf16_t* __restrict__ Q, const bf16_t* __restrict__ K,
             const bf16_t* __restrict__ Vt, bf16_t* __restrict__ out) {
  __shared__ bf16_t Ssc[128 * 136];  // masked scores tile (bf16), pitch 136

  const int t    = threadIdx.x;
  const int lane = t & 31;
  const int wave = t >> 5;
  const int wm   = wave >> 1;
  const int wn   = wave & 1;
  const int l15  = lane & 15;
  const int hi   = (lane >= 16);
  const int aoff = hi ? 16 : 0;
  const int boff = hi ? 16 : 0;
  const int rb   = hi ? 8 : 0;

  const int band = blockIdx.x;       // 0..15
  const int bh   = blockIdx.y;       // 0..31
  const int bb   = bh >> 4;
  const int h    = bh & 15;
  const int t0   = band * 128;

  const bf16_t* Qh = Q  + (size_t)bh * SEQ * DHEAD;
  const bf16_t* Kh = K  + (size_t)bh * SEQ * DHEAD;
  const bf16_t* Vh = Vt + (size_t)bh * DHEAD * SEQ;

  // Q fragments register-resident for the whole band: [m-subtile][k-step]
  v16bf qf[2][2];
#pragma unroll
  for (int m2 = 0; m2 < 2; ++m2)
#pragma unroll
    for (int ks = 0; ks < 2; ++ks)
      qf[m2][ks] = load_afrag(
          Qh + (size_t)(t0 + wm * 32 + m2 * 16 + l15) * DHEAD + ks * 32 + aoff);

  v8f oacc[4];
#pragma unroll
  for (int i = 0; i < 4; ++i) oacc[i] = zero8();

  for (int st = 0; st <= band; ++st) {   // causal: only s-tiles <= band
    const int s0 = st * 128;

    v8f sc[2][4];
#pragma unroll
    for (int m2 = 0; m2 < 2; ++m2)
#pragma unroll
      for (int i = 0; i < 4; ++i) sc[m2][i] = zero8();

    // scores = Q * K^T  (128x128, k=64)
#pragma unroll
    for (int ks = 0; ks < 2; ++ks) {
#pragma unroll
      for (int i = 0; i < 4; ++i) {
        v16bf kf = load_contig16(
            Kh + (size_t)(s0 + wn * 64 + i * 16 + l15) * DHEAD + ks * 32 + boff);
#pragma unroll
        for (int m2 = 0; m2 < 2; ++m2)
          sc[m2][i] = __builtin_amdgcn_wmma_f32_16x16x32_bf16(
              false, qf[m2][ks], false, kf, (short)0, sc[m2][i], false, false);
      }
    }

    // causal mask on diagonal tile + round-trip through LDS as bf16 A-operand
    const bool diag = (st == band);
#pragma unroll
    for (int m2 = 0; m2 < 2; ++m2) {
#pragma unroll
      for (int i = 0; i < 4; ++i) {
        const int sl = wn * 64 + i * 16 + l15;
#pragma unroll
        for (int r = 0; r < 8; ++r) {
          const int ml = wm * 32 + m2 * 16 + rb + r;
          float v = sc[m2][i][r];
          if (diag && sl > ml) v = 0.0f;
          Ssc[ml * 136 + sl] = (bf16_t)v;
        }
      }
    }
    __syncthreads();

    // out += scores * V   (each wave: 16 rows x 64 cols, k=128)
#pragma unroll
    for (int ks2 = 0; ks2 < 4; ++ks2) {
      v16bf af = load_afrag(&Ssc[(wave * 16 + l15) * 136 + ks2 * 32 + aoff]);
#pragma unroll
      for (int i = 0; i < 4; ++i) {
        v16bf vf = load_contig16(
            Vh + (size_t)(i * 16 + l15) * SEQ + s0 + ks2 * 32 + boff);
        oacc[i] = __builtin_amdgcn_wmma_f32_16x16x32_bf16(
            false, af, false, vf, (short)0, oacc[i], false, false);
      }
    }
    __syncthreads();
  }

  // concat-heads store: attn[b][l][h*64+d], bf16
#pragma unroll
  for (int i = 0; i < 4; ++i) {
    const int d = i * 16 + l15;
#pragma unroll
    for (int r = 0; r < 8; ++r) {
      const int lrow = t0 + wave * 16 + rb + r;
      out[(size_t)(bb * SEQ + lrow) * DMODEL + h * DHEAD + d] = (bf16_t)oacc[i][r];
    }
  }
}

// ---------------------------------------------------------------- launch
extern "C" void kernel_launch(void* const* d_in, const int* in_sizes, int n_in,
                              void* d_out, int out_size, void* d_ws, size_t ws_size,
                              hipStream_t stream) {
  (void)in_sizes; (void)n_in; (void)out_size; (void)ws_size;

  const float* x  = (const float*)d_in[0];
  const float* wq = (const float*)d_in[1];
  const float* bq = (const float*)d_in[2];
  const float* wk = (const float*)d_in[3];
  const float* bk = (const float*)d_in[4];
  const float* wv = (const float*)d_in[5];
  const float* bv = (const float*)d_in[6];
  const float* wo = (const float*)d_in[7];
  const float* bo = (const float*)d_in[8];

  const size_t SZ_X = (size_t)MTOT * DMODEL;     // 4M elements
  const size_t SZ_W = (size_t)DMODEL * DMODEL;   // 1M elements

  char* ws = (char*)d_ws;
  bf16_t* xb   = (bf16_t*)ws; ws += SZ_X * 2;
  bf16_t* wqb  = (bf16_t*)ws; ws += SZ_W * 2;
  bf16_t* wkb  = (bf16_t*)ws; ws += SZ_W * 2;
  bf16_t* wvb  = (bf16_t*)ws; ws += SZ_W * 2;
  bf16_t* wob  = (bf16_t*)ws; ws += SZ_W * 2;
  bf16_t* Qb   = (bf16_t*)ws; ws += SZ_X * 2;
  bf16_t* Kb   = (bf16_t*)ws; ws += SZ_X * 2;
  bf16_t* Vtb  = (bf16_t*)ws; ws += SZ_X * 2;
  bf16_t* attn = (bf16_t*)ws; ws += SZ_X * 2;    // ~48 MiB total

  // fp32 -> bf16 conversions
  {
    const int nx4 = (int)(SZ_X / 4), nw4 = (int)(SZ_W / 4);
    cvt_f32_bf16<<<(nx4 + 255) / 256, 256, 0, stream>>>((const float4*)x,  (bf16x4*)xb,  nx4);
    cvt_f32_bf16<<<(nw4 + 255) / 256, 256, 0, stream>>>((const float4*)wq, (bf16x4*)wqb, nw4);
    cvt_f32_bf16<<<(nw4 + 255) / 256, 256, 0, stream>>>((const float4*)wk, (bf16x4*)wkb, nw4);
    cvt_f32_bf16<<<(nw4 + 255) / 256, 256, 0, stream>>>((const float4*)wv, (bf16x4*)wvb, nw4);
    cvt_f32_bf16<<<(nw4 + 255) / 256, 256, 0, stream>>>((const float4*)wo, (bf16x4*)wob, nw4);
  }

  const dim3 ggrid(DMODEL / 128, MTOT / 128);    // (8, 32)
  gemm_bf16<0><<<ggrid, 256, 0, stream>>>(xb, wqb, bq, (void*)Qb);
  gemm_bf16<0><<<ggrid, 256, 0, stream>>>(xb, wkb, bk, (void*)Kb);
  gemm_bf16<1><<<ggrid, 256, 0, stream>>>(xb, wvb, bv, (void*)Vtb);

  linattn<<<dim3(SEQ / 128, BATCH * NHEADS), 256, 0, stream>>>(Qb, Kb, Vtb, attn);

  gemm_bf16<2><<<ggrid, 256, 0, stream>>>(attn, wob, bo, d_out);
}
